// SAGPoolG_57784490000619
// MI455X (gfx1250) — matrix operand order repaired and checked by reading, not compiled, not run on hardware
//
#include <hip/hip_runtime.h>
#include <hip/hip_bf16.h>
#include <math.h>

// ---------------- problem constants (match reference) ----------------
#define G_    128          // graphs
#define NPG_  64           // nodes per graph
#define NN_   8192         // total nodes
#define H_    128          // hidden
#define F3_   384          // 3*H concat feature width
#define EPG_  1024         // edges per graph (NPG*DEG)
#define EDGES_ (NN_*16)
#define K_    32           // top-k keep per graph
#define DOUT_ 10

typedef __attribute__((ext_vector_type(16))) __bf16 v16bf;
typedef __attribute__((ext_vector_type(8)))  float  v8f;

// ---------------- WMMA fragment loaders (CDNA5 wave32 layouts) ----------------
// A operand: 16x32 bf16 tile, row-major source [M][K] with leading dim ld.
// lane l: row m = l&15; elements 0..15 -> K = 16*(l>>4) + e
__device__ __forceinline__ v16bf load_a_frag(const __bf16* base, int ld, int lane) {
  const int m  = lane & 15;
  const int k0 = (lane >> 4) << 4;           // 0 or 16
  const __bf16* p0 = base + m * ld + k0;
  v16bf r;
#pragma unroll
  for (int i = 0; i < 16; ++i) { r[i] = p0[i]; }
  return r;
}

// B operand: 32x16 bf16 tile, source is B^T row-major [N][K] with leading dim ld
// (i.e. column n of B is contiguous over K). lane l: col n = l&15,
// K = 16*(l>>4) + e for e = 0..15  -> one contiguous 32-byte read.
__device__ __forceinline__ v16bf load_b_frag(const __bf16* base, int ld, int lane) {
  const int n  = lane & 15;
  const int k0 = (lane >> 4) << 4;           // 0 or 16
  const __bf16* p = base + n * ld + k0;
  v16bf r;
#pragma unroll
  for (int i = 0; i < 16; ++i) r[i] = p[i];
  return r;
}

// ---------------- prep kernels ----------------
__global__ void cvt_f32_bf16_kernel(const float* __restrict__ in,
                                    __bf16* __restrict__ out, int n) {
  int i = blockIdx.x * blockDim.x + threadIdx.x;
  if (i < n) out[i] = (__bf16)in[i];
}

// W is [K=128][N=128] row-major (in_dim major); produce Wt[n][k] bf16.
__global__ void transpose_w_kernel(const float* __restrict__ W,
                                   __bf16* __restrict__ Wt) {
  int i = blockIdx.x * blockDim.x + threadIdx.x;
  if (i < H_ * H_) {
    int n = i >> 7, k = i & 127;
    Wt[i] = (__bf16)W[k * H_ + n];
  }
}

// ---------------- per-graph normalized adjacency (block-diagonal) ----------------
__global__ __launch_bounds__(256)
void adj_kernel(const int* __restrict__ ei,       // [2][EDGES_] src row then dst row
                float*  __restrict__ Af,          // [G][64][64] f32 (for score path)
                __bf16* __restrict__ Abf) {       // [G][64][64] bf16 (for WMMA)
  __shared__ float A[NPG_ * NPG_];
  __shared__ float dinv[NPG_];
  const int g = blockIdx.x, t = threadIdx.x;

  for (int i = t; i < NPG_ * NPG_; i += 256) A[i] = 0.0f;
  __syncthreads();

  const int* srcp = ei;
  const int* dstp = ei + EDGES_;
  const int base = g * EPG_;                      // edges are grouped per graph
  for (int e = t; e < EPG_; e += 256) {
    int sl = srcp[base + e] & (NPG_ - 1);
    int dl = dstp[base + e] & (NPG_ - 1);
    A[sl * NPG_ + dl] = 1.0f;                     // symmetrize: max(A, A^T)
    A[dl * NPG_ + sl] = 1.0f;
  }
  __syncthreads();
  if (t < NPG_) A[t * NPG_ + t] += 1.0f;          // + I (after symmetrize)
  __syncthreads();
  if (t < NPG_) {
    float sum = 0.0f;
#pragma unroll 8
    for (int j = 0; j < NPG_; ++j) sum += A[t * NPG_ + j];
    dinv[t] = rsqrtf(sum);
  }
  __syncthreads();
  for (int i = t; i < NPG_ * NPG_; i += 256) {
    int r = i >> 6, c = i & (NPG_ - 1);
    float v = A[i] * dinv[r] * dinv[c];
    Af [(size_t)g * NPG_ * NPG_ + i] = v;
    Abf[(size_t)g * NPG_ * NPG_ + i] = (__bf16)v;
  }
}

// ---------------- fused GCN layer: H = relu(A_g @ (X_g @ W) + b), per graph ----------------
// 4 waves; wave w owns rows [16w, 16w+16), 8 N-tiles of 16.
__global__ __launch_bounds__(128)
void gcn_kernel(const __bf16* __restrict__ Xin, int xStride,   // [8192][xStride]
                const __bf16* __restrict__ Wt,                 // [128][128] = W^T bf16
                const float*  __restrict__ bias,               // [128]
                const __bf16* __restrict__ Abf,                // [G][64][64]
                __bf16* __restrict__ Hout, int hStride) {      // [8192][hStride]
  __shared__ __align__(16) __bf16 Xs[NPG_ * H_];     // 16 KB : graph input tile
  __shared__ __align__(16) __bf16 As[NPG_ * NPG_];   //  8 KB : adjacency block
  __shared__ __align__(16) __bf16 Yt[H_ * NPG_];     // 16 KB : Y^T (B-operand layout)

  const int g    = blockIdx.x;
  const int tid  = threadIdx.x;
  const int lane = tid & 31;
  const int w    = tid >> 5;                  // wave id 0..3 -> M tile

  // stage X_g and A_g into LDS
  for (int i = tid; i < NPG_ * H_; i += 128) {
    int r = i >> 7, c = i & 127;
    Xs[i] = Xin[(size_t)(g * NPG_ + r) * xStride + c];
  }
  const __bf16* Ag = Abf + (size_t)g * NPG_ * NPG_;
  for (int i = tid; i < NPG_ * NPG_; i += 128) As[i] = Ag[i];
  __syncthreads();

  // ---- GEMM1: Y = Xs(64x128) @ W(128x128), K = 128 in 4 WMMA steps ----
  v8f acc[8];
#pragma unroll
  for (int nt = 0; nt < 8; ++nt) acc[nt] = (v8f){};
#pragma unroll
  for (int kt = 0; kt < 4; ++kt) {
    v16bf afrag = load_a_frag(&Xs[(w * 16) * H_ + kt * 32], H_, lane);
#pragma unroll
    for (int nt = 0; nt < 8; ++nt) {
      v16bf bfrag = load_b_frag(&Wt[(nt * 16) * H_ + kt * 32], H_, lane);
      acc[nt] = __builtin_amdgcn_wmma_f32_16x16x32_bf16(
          false, afrag, false, bfrag, (short)0, acc[nt], false, false);
    }
  }

  // store Y transposed into LDS as bf16 (column-contiguous => B-frag friendly)
  {
    const int n_l = lane & 15;
    const int m_b = w * 16 + ((lane >> 4) << 3);
#pragma unroll
    for (int nt = 0; nt < 8; ++nt) {
      const int n0 = nt * 16 + n_l;
#pragma unroll
      for (int r = 0; r < 8; ++r)
        Yt[n0 * NPG_ + m_b + r] = (__bf16)acc[nt][r];
    }
  }
  __syncthreads();

  // ---- GEMM2: Z = A_g(64x64) @ Y(64x128), K = 64 in 2 WMMA steps ----
  v8f acc2[8];
#pragma unroll
  for (int nt = 0; nt < 8; ++nt) acc2[nt] = (v8f){};
#pragma unroll
  for (int kt = 0; kt < 2; ++kt) {
    v16bf afrag = load_a_frag(&As[(w * 16) * NPG_ + kt * 32], NPG_, lane);
#pragma unroll
    for (int nt = 0; nt < 8; ++nt) {
      v16bf bfrag = load_b_frag(&Yt[(nt * 16) * NPG_ + kt * 32], NPG_, lane);
      acc2[nt] = __builtin_amdgcn_wmma_f32_16x16x32_bf16(
          false, afrag, false, bfrag, (short)0, acc2[nt], false, false);
    }
  }

  // ---- bias + relu + store bf16 ----
  {
    const int n_l = lane & 15;
    const int m_b = w * 16 + ((lane >> 4) << 3);
#pragma unroll
    for (int nt = 0; nt < 8; ++nt) {
      const int n0 = nt * 16 + n_l;
      const float bn = bias[n0];
#pragma unroll
      for (int r = 0; r < 8; ++r) {
        float v = fmaxf(acc2[nt][r] + bn, 0.0f);
        Hout[(size_t)(g * NPG_ + m_b + r) * hStride + n0] = (__bf16)v;
      }
    }
  }
}

// ---------------- attention score + top-K + mean||max readout (per graph) ----------------
__global__ __launch_bounds__(128)
void pool_kernel(const __bf16* __restrict__ feats,   // [8192][384]
                 const float*  __restrict__ Af,      // [G][64][64]
                 const float*  __restrict__ Wa,      // [384]
                 const float*  __restrict__ ba,      // [1]
                 float* __restrict__ ro) {           // [G][768]
  __shared__ float p[NPG_];
  __shared__ float s[NPG_];
  __shared__ int   sel[NPG_];
  const int g = blockIdx.x, t = threadIdx.x;

  if (t < NPG_) {                       // p = feats @ Wa
    const __bf16* f = feats + (size_t)(g * NPG_ + t) * F3_;
    float acc = 0.0f;
#pragma unroll 8
    for (int d = 0; d < F3_; ++d) acc += (float)f[d] * Wa[d];
    p[t] = acc;
  }
  __syncthreads();
  if (t < NPG_) {                       // s = tanh(A @ p + ba)
    const float* Ar = Af + (size_t)g * NPG_ * NPG_ + t * NPG_;
    float acc = 0.0f;
#pragma unroll 8
    for (int j = 0; j < NPG_; ++j) acc += Ar[j] * p[j];
    s[t] = tanhf(acc + ba[0]);
  }
  __syncthreads();
  if (t < NPG_) {                       // rank with top_k tie-break (lower idx first)
    const float st = s[t];
    int rank = 0;
    for (int j = 0; j < NPG_; ++j) {
      float sj = s[j];
      rank += (sj > st) || (sj == st && j < t);
    }
    sel[t] = (rank < K_) ? 1 : 0;
  }
  __syncthreads();
  for (int d = t; d < F3_; d += 128) {  // masked mean || max over selected set
    float sum = 0.0f, mx = -INFINITY;
    for (int j = 0; j < NPG_; ++j) {
      if (sel[j]) {
        float v = (float)feats[(size_t)(g * NPG_ + j) * F3_ + d] * s[j];
        sum += v;
        mx = fmaxf(mx, v);
      }
    }
    ro[(size_t)g * (2 * F3_) + d]        = sum * (1.0f / (float)K_);
    ro[(size_t)g * (2 * F3_) + F3_ + d]  = mx;
  }
}

// ---------------- final MLP (f32, per graph) ----------------
__global__ __launch_bounds__(128)
void mlp_kernel(const float* __restrict__ ro,                     // [G][768]
                const float* __restrict__ M1, const float* __restrict__ c1,
                const float* __restrict__ M2, const float* __restrict__ c2,
                const float* __restrict__ M3, const float* __restrict__ c3,
                float* __restrict__ out) {                        // [G][10]
  __shared__ float h1[H_];
  __shared__ float h2[H_ / 2];
  const int g = blockIdx.x, t = threadIdx.x;
  const float* r = ro + (size_t)g * (2 * F3_);
  {
    float acc = c1[t];
#pragma unroll 8
    for (int d = 0; d < 2 * F3_; ++d) acc += r[d] * M1[d * H_ + t];
    h1[t] = fmaxf(acc, 0.0f);
  }
  __syncthreads();
  if (t < H_ / 2) {
    float acc = c2[t];
#pragma unroll 8
    for (int k = 0; k < H_; ++k) acc += h1[k] * M2[k * (H_ / 2) + t];
    h2[t] = fmaxf(acc, 0.0f);
  }
  __syncthreads();
  if (t < DOUT_) {
    float acc = c3[t];
#pragma unroll 8
    for (int k = 0; k < H_ / 2; ++k) acc += h2[k] * M3[k * DOUT_ + t];
    out[g * DOUT_ + t] = acc;
  }
}

// ---------------- launcher ----------------
extern "C" void kernel_launch(void* const* d_in, const int* in_sizes, int n_in,
                              void* d_out, int out_size, void* d_ws, size_t ws_size,
                              hipStream_t stream) {
  const float* x  = (const float*)d_in[0];
  const int*   ei = (const int*)d_in[1];
  // d_in[2] = batch: structure is implicit (contiguous 64-node blocks) -> unused
  const float* W1 = (const float*)d_in[3];  const float* b1 = (const float*)d_in[4];
  const float* W2 = (const float*)d_in[5];  const float* b2 = (const float*)d_in[6];
  const float* W3 = (const float*)d_in[7];  const float* b3 = (const float*)d_in[8];
  const float* Wa = (const float*)d_in[9];  const float* ba = (const float*)d_in[10];
  const float* M1 = (const float*)d_in[11]; const float* c1 = (const float*)d_in[12];
  const float* M2 = (const float*)d_in[13]; const float* c2 = (const float*)d_in[14];
  const float* M3 = (const float*)d_in[15]; const float* c3 = (const float*)d_in[16];

  // workspace layout (bytes)
  char* ws = (char*)d_ws;
  size_t off = 0;
  auto take = [&](size_t bytes) { char* p = ws + off; off = (off + bytes + 255) & ~(size_t)255; return p; };
  __bf16* Xbf  = (__bf16*)take((size_t)NN_ * H_ * 2);            // 2 MB
  __bf16* Wt1  = (__bf16*)take((size_t)H_ * H_ * 2);
  __bf16* Wt2  = (__bf16*)take((size_t)H_ * H_ * 2);
  __bf16* Wt3  = (__bf16*)take((size_t)H_ * H_ * 2);
  __bf16* Abf  = (__bf16*)take((size_t)G_ * NPG_ * NPG_ * 2);    // 1 MB
  float*  Af   = (float*) take((size_t)G_ * NPG_ * NPG_ * 4);    // 2 MB
  __bf16* feat = (__bf16*)take((size_t)NN_ * F3_ * 2);           // 6 MB
  float*  ro   = (float*) take((size_t)G_ * 2 * F3_ * 4);        // 384 KB
  (void)ws_size; (void)in_sizes; (void)n_in; (void)out_size;

  // prep: bf16 copies / transposed weights, and block-diagonal normalized A
  cvt_f32_bf16_kernel<<<(NN_ * H_ + 255) / 256, 256, 0, stream>>>(x, Xbf, NN_ * H_);
  transpose_w_kernel<<<(H_ * H_ + 255) / 256, 256, 0, stream>>>(W1, Wt1);
  transpose_w_kernel<<<(H_ * H_ + 255) / 256, 256, 0, stream>>>(W2, Wt2);
  transpose_w_kernel<<<(H_ * H_ + 255) / 256, 256, 0, stream>>>(W3, Wt3);
  adj_kernel<<<G_, 256, 0, stream>>>(ei, Af, Abf);

  // 3 fused GCN layers writing into the concatenated feats buffer [N][384]
  gcn_kernel<<<G_, 128, 0, stream>>>(Xbf, H_,        Wt1, b1, Abf, feat + 0 * H_, F3_);
  gcn_kernel<<<G_, 128, 0, stream>>>(feat + 0 * H_, F3_, Wt2, b2, Abf, feat + 1 * H_, F3_);
  gcn_kernel<<<G_, 128, 0, stream>>>(feat + 1 * H_, F3_, Wt3, b3, Abf, feat + 2 * H_, F3_);

  // attention + top-K + readout, then MLP head
  pool_kernel<<<G_, 128, 0, stream>>>(feat, Af, Wa, ba, ro);
  mlp_kernel<<<G_, 128, 0, stream>>>(ro, M1, c1, M2, c2, M3, c3, (float*)d_out);
}
